// Model4_84774064488932
// MI455X (gfx1250) — compile-verified, matrix operand order checked by reading, not run on hardware
//
#include <hip/hip_runtime.h>
#include <hip/hip_bf16.h>

// ---------------------------------------------------------------------------
// MI455X (gfx1250): bf16 WMMA pipeline with fp32 accumulation.
// NT GEMM: 32x64 wave tile (8 wmma / K-step). NN GEMM: TDM async B-tile to LDS.
// ---------------------------------------------------------------------------

typedef __bf16 bf16;
typedef __attribute__((ext_vector_type(16))) __bf16 v16bf;
typedef __attribute__((ext_vector_type(8)))  __bf16 v8bf;
typedef __attribute__((ext_vector_type(8)))  float  v8f;
typedef __attribute__((ext_vector_type(4)))  unsigned int v4u;
typedef __attribute__((ext_vector_type(8)))  int v8i;
typedef __attribute__((ext_vector_type(4)))  int v4i;

#define EDIM  1024
#define NHEAD 4
#define DHEAD 256

#ifndef __has_builtin
#define __has_builtin(x) 0
#endif
#if __has_builtin(__builtin_amdgcn_tensor_load_to_lds)
#define HAVE_TDM 1
#if __has_include(<hip/amd_detail/amd_gfx1250_TDM.h>)
#define TDM_6ARG 1
#endif
#endif

__device__ __forceinline__ bf16 f2bf(float f) {
    unsigned u = __builtin_bit_cast(unsigned, f);
    unsigned r = u + 0x7fffu + ((u >> 16) & 1u);   // round-to-nearest-even
    unsigned short h = (unsigned short)(r >> 16);
    return __builtin_bit_cast(bf16, h);
}

__device__ __forceinline__ v16bf cat8(v8bf lo, v8bf hi) {
    v16bf r;
#pragma unroll
    for (int i = 0; i < 8; i++) { r[i] = lo[i]; r[8 + i] = hi[i]; }
    return r;
}

__device__ __forceinline__ v8f wmma_bf(v16bf a, v16bf b, v8f c) {
    return __builtin_amdgcn_wmma_f32_16x16x32_bf16(false, a, false, b, (short)0, c,
                                                   false, false);
}

// ---------------------------------------------------------------------------
// NT GEMM: C[b] = alpha * A[b](MxK) * B[b](NxK)^T + bias + resid
// Wave tile 32(M) x 64(N) = 2x4 subtiles. Block = 8 waves = 64(M) x 256(N).
// ---------------------------------------------------------------------------
__global__ void __launch_bounds__(256) gemm_nt(
    const bf16* __restrict__ A, long sA, int lda,
    const bf16* __restrict__ B, long sB, int ldb,
    float* __restrict__ Cf, bf16* __restrict__ Cb, long sC, int ldc,
    const float* __restrict__ bias, const float* __restrict__ resid,
    int K, float alpha)
{
    const int lane = threadIdx.x & 31;
    const int wave = threadIdx.x >> 5;
    const int l15  = lane & 15;
    const int half = lane >> 4;
    const int m0 = blockIdx.x * 64 + (wave & 1) * 32;
    const int n0 = blockIdx.y * 256 + (wave >> 1) * 64;
    const int b  = blockIdx.z;

    const bf16* ap0 = A + (long)b * sA + (long)(m0 + l15) * lda + half * 8;
    const bf16* ap1 = ap0 + 16L * lda;
    const bf16* bp0 = B + (long)b * sB + (long)(n0 + l15) * ldb + half * 8;
    const long  bstep = 16L * ldb;

    v8f acc[2][4] = {};
    for (int k = 0; k < K; k += 32) {
        __builtin_prefetch(ap0 + k + 256, 0, 3);   // global_prefetch_b8
        __builtin_prefetch(ap1 + k + 256, 0, 3);
        __builtin_prefetch(bp0 + k + 256, 0, 3);
        __builtin_prefetch(bp0 + 2 * bstep + k + 256, 0, 3);

        v16bf a0 = cat8(*(const v8bf*)(ap0 + k), *(const v8bf*)(ap0 + k + 16));
        v16bf a1 = cat8(*(const v8bf*)(ap1 + k), *(const v8bf*)(ap1 + k + 16));
        v16bf bv[4];
#pragma unroll
        for (int j = 0; j < 4; j++) {
            const bf16* bp = bp0 + j * bstep + k;
            bv[j] = cat8(*(const v8bf*)(bp), *(const v8bf*)(bp + 16));
        }
#pragma unroll
        for (int j = 0; j < 4; j++) {
            acc[0][j] = wmma_bf(a0, bv[j], acc[0][j]);
            acc[1][j] = wmma_bf(a1, bv[j], acc[1][j]);
        }
    }

#pragma unroll
    for (int mi = 0; mi < 2; mi++) {
#pragma unroll
        for (int j = 0; j < 4; j++) {
            const int n = n0 + j * 16 + l15;
            const float bb = bias ? bias[n] : 0.0f;
#pragma unroll
            for (int v = 0; v < 8; v++) {
                const int m = m0 + mi * 16 + half * 8 + v;
                const long off = (long)b * sC + (long)m * ldc + n;
                float val = acc[mi][j][v] * alpha + bb;
                if (resid) val += resid[(long)m * ldc + n];  // only used when batch==1
                if (Cf) Cf[off] = val;
                if (Cb) Cb[off] = f2bf(val);
            }
        }
    }
}

// ---------------------------------------------------------------------------
// NN GEMM (attn @ V): per-wave 64x64 bf16 B tile DMA'd into LDS by the
// Tensor Data Mover (tensor_load_to_lds, TENSORcnt), columns read via DS.
// Wave tile 32(M) x 64(N); block = 8 waves = 64(M) x 256(N).
// ---------------------------------------------------------------------------
__global__ void __launch_bounds__(256) gemm_nn(
    const bf16* __restrict__ A, long sA, int lda,
    const bf16* __restrict__ B, long sB, int ldb,
    float* __restrict__ Cf, bf16* __restrict__ Cb, long sC, int ldc,
    int K, float alpha)
{
    __shared__ __align__(16) bf16 smem[8][64 * 64];   // 8 KB per wave
    const int lane = threadIdx.x & 31;
    const int wave = threadIdx.x >> 5;
    const int l15  = lane & 15;
    const int half = lane >> 4;
    const int m0 = blockIdx.x * 64 + (wave & 1) * 32;
    const int n0 = blockIdx.y * 256 + (wave >> 1) * 64;
    const int b  = blockIdx.z;

    const bf16* ap0 = A + (long)b * sA + (long)(m0 + l15) * lda + half * 8;
    const bf16* ap1 = ap0 + 16L * lda;
    const bf16* Bb  = B + (long)b * sB + n0;          // row k lives at + k*ldb
    bf16* lds = &smem[wave][0];
    const unsigned lds_off = (unsigned)(size_t)lds;   // low 32 bits = LDS byte offset

    v8f acc[2][4] = {};
    for (int k = 0; k < K; k += 64) {
#if defined(HAVE_TDM) && defined(__AMDGCN__)
        // Previous iteration's DS reads must land before the TDM engine
        // (unordered with wave DS ops) overwrites this wave's LDS slice.
        asm volatile("s_wait_dscnt 0x0" ::: "memory");
        // ---- Tensor DMA descriptor (ISA 8.3/8.4): 64x64 bf16 tile ----
        const unsigned long long ga = (unsigned long long)(size_t)(Bb + (long)k * ldb);
        v4u g0;
        g0[0] = 1u;                                   // count=1, user mode
        g0[1] = lds_off;                              // lds_addr (bytes)
        g0[2] = (unsigned)ga;                         // global_addr[31:0]
        g0[3] = (unsigned)((ga >> 32) & 0x01FFFFFFu) | (2u << 30);  // addr[56:32] | type=2
        const unsigned td0 = 1u << 20, td1 = 1u << 20;    // generous OOB dims
        v8i g1;
        g1[0] = (int)(1u << 16);                      // data_size=1 (2 bytes)
        g1[1] = (int)((td0 & 0xffffu) << 16);         // tensor_dim0 lo16 (atomic addr = 0)
        g1[2] = (int)(((td0 >> 16) & 0xffffu) | ((td1 & 0xffffu) << 16));
        g1[3] = (int)(((td1 >> 16) & 0xffffu) | (64u << 16));   // tile_dim0 = 64
        g1[4] = (int)64u;                             // tile_dim1 = 64, tile_dim2 = 0
        g1[5] = (int)(unsigned)ldb;                   // tensor_dim0_stride (elements)
        g1[6] = 0; g1[7] = 0;
        v4i gz; gz[0] = 0; gz[1] = 0; gz[2] = 0; gz[3] = 0;
#if defined(TDM_6ARG)
        v8i gz8; gz8[0]=0; gz8[1]=0; gz8[2]=0; gz8[3]=0; gz8[4]=0; gz8[5]=0; gz8[6]=0; gz8[7]=0;
        __builtin_amdgcn_tensor_load_to_lds(g0, g1, gz, gz, gz8, 0);
#else
        __builtin_amdgcn_tensor_load_to_lds(g0, g1, gz, gz, 0);
#endif
#if __has_builtin(__builtin_amdgcn_s_wait_tensorcnt)
        __builtin_amdgcn_s_wait_tensorcnt(0);
#else
        asm volatile("s_wait_tensorcnt 0x0" ::: "memory");
#endif
#else
        // Fallback: cooperative loads (per-wave LDS ops are in order).
        for (int r = lane; r < 64; r += 32) {
            const bf16* brow = Bb + (long)(k + r) * ldb;
#pragma unroll
            for (int c = 0; c < 64; c += 8)
                *(v8bf*)&lds[r * 64 + c] = *(const v8bf*)(brow + c);
        }
#endif
        __builtin_prefetch(ap0 + k + 256, 0, 3);
        __builtin_prefetch(ap1 + k + 256, 0, 3);
#pragma unroll
        for (int kk = 0; kk < 64; kk += 32) {
            v16bf a0 = cat8(*(const v8bf*)(ap0 + k + kk), *(const v8bf*)(ap0 + k + kk + 16));
            v16bf a1 = cat8(*(const v8bf*)(ap1 + k + kk), *(const v8bf*)(ap1 + k + kk + 16));
#pragma unroll
            for (int j = 0; j < 4; j++) {
                v16bf bvv;
#pragma unroll
                for (int i = 0; i < 8; i++) {
                    bvv[i]     = lds[(kk + half * 8 + i) * 64 + j * 16 + l15];
                    bvv[8 + i] = lds[(kk + 16 + half * 8 + i) * 64 + j * 16 + l15];
                }
                acc[0][j] = wmma_bf(a0, bvv, acc[0][j]);
                acc[1][j] = wmma_bf(a1, bvv, acc[1][j]);
            }
        }
    }

#pragma unroll
    for (int mi = 0; mi < 2; mi++) {
#pragma unroll
        for (int j = 0; j < 4; j++) {
            const int n = n0 + j * 16 + l15;
#pragma unroll
            for (int v = 0; v < 8; v++) {
                const int m = m0 + mi * 16 + half * 8 + v;
                const long off = (long)b * sC + (long)m * ldc + n;
                float val = acc[mi][j][v] * alpha;
                if (Cf) Cf[off] = val;
                if (Cb) Cb[off] = f2bf(val);
            }
        }
    }
}

// ---------------------------------------------------------------------------
// Row softmax: one block per row, fp32 in -> bf16 probabilities out.
// ---------------------------------------------------------------------------
__global__ void __launch_bounds__(256) softmax_rows(
    const float* __restrict__ S, bf16* __restrict__ P, int n)
{
    __shared__ float red[256];
    const long row = blockIdx.x;
    const float* s = S + row * (long)n;
    bf16* p = P + row * (long)n;
    const int tid = threadIdx.x;

    float mx = -3.4e38f;
    for (int c = tid; c < n; c += 256) mx = fmaxf(mx, s[c]);
    red[tid] = mx; __syncthreads();
    for (int k = 128; k > 0; k >>= 1) { if (tid < k) red[tid] = fmaxf(red[tid], red[tid + k]); __syncthreads(); }
    mx = red[0]; __syncthreads();

    float sum = 0.0f;
    for (int c = tid; c < n; c += 256) sum += __expf(s[c] - mx);
    red[tid] = sum; __syncthreads();
    for (int k = 128; k > 0; k >>= 1) { if (tid < k) red[tid] += red[tid + k]; __syncthreads(); }
    const float inv = 1.0f / red[0];

    for (int c = tid; c < n; c += 256) p[c] = f2bf(__expf(s[c] - mx) * inv);
}

__device__ __forceinline__ float blk_sumsq(const float* x, int n, float* red) {
    const int tid = threadIdx.x;
    float s = 0.0f;
    for (int c = tid; c < n; c += 256) { float v = x[c]; s += v * v; }
    red[tid] = s; __syncthreads();
    for (int k = 128; k > 0; k >>= 1) { if (tid < k) red[tid] += red[tid + k]; __syncthreads(); }
    float r = red[0]; __syncthreads();
    return r;
}

// prep: l2-normalize local/global/text rows; emit bf16 streams (+pos embeds).
__global__ void __launch_bounds__(256) prep(
    const float* __restrict__ lf, const float* __restrict__ gf, const float* __restrict__ tf,
    const float* __restrict__ posl, const float* __restrict__ posg,
    bf16* __restrict__ localn, bf16* __restrict__ klocal,
    bf16* __restrict__ kglobal, bf16* __restrict__ textn)
{
    __shared__ float red[256];
    const long r = blockIdx.x;
    const int tid = threadIdx.x;

    const float* x = lf + r * EDIM;
    float inv = rsqrtf(blk_sumsq(x, EDIM, red));
    for (int c = tid; c < EDIM; c += 256) {
        float v = x[c] * inv;
        localn[r * EDIM + c] = f2bf(v);
        klocal[r * EDIM + c] = f2bf(v + posl[c]);
    }
    x = gf + r * EDIM;
    inv = rsqrtf(blk_sumsq(x, EDIM, red));
    for (int c = tid; c < EDIM; c += 256)
        kglobal[r * EDIM + c] = f2bf(x[c] * inv + posg[c]);

    x = tf + r * EDIM;
    inv = rsqrtf(blk_sumsq(x, EDIM, red));
    for (int c = tid; c < EDIM; c += 256)
        textn[r * EDIM + c] = f2bf(x[c] * inv);
}

// rownorm with EPS clamp (for 'full' -> ff_n)
__global__ void __launch_bounds__(256) rownorm(const float* __restrict__ X, bf16* __restrict__ Y)
{
    __shared__ float red[256];
    const long r = blockIdx.x;
    const float* x = X + r * EDIM;
    float ss = blk_sumsq(x, EDIM, red);
    float inv = 1.0f / fmaxf(sqrtf(ss), 1e-8f);
    for (int c = threadIdx.x; c < EDIM; c += 256) Y[r * EDIM + c] = f2bf(x[c] * inv);
}

__global__ void __launch_bounds__(256) f2b_k(const float* __restrict__ x, bf16* __restrict__ y, int n)
{
    int i = blockIdx.x * 256 + threadIdx.x;
    if (i < n) y[i] = f2bf(x[i]);
}

// ---------------------------------------------------------------------------
// Host orchestration
// ---------------------------------------------------------------------------
extern "C" void kernel_launch(void* const* d_in, const int* in_sizes, int n_in,
                              void* d_out, int out_size, void* d_ws, size_t ws_size,
                              hipStream_t stream)
{
    (void)in_sizes; (void)n_in; (void)out_size; (void)ws_size;
    const long E2 = (long)EDIM * EDIM;

    const float* local_feat  = (const float*)d_in[0];
    const float* global_feat = (const float*)d_in[1];
    const float* text_feat   = (const float*)d_in[2];
    // d_in[3] = n_groups (device scalar); fixed G=4 per reference setup
    const float* w_tl = (const float*)d_in[4];  const float* b_tl = (const float*)d_in[5];
    const float* w_tg = (const float*)d_in[6];  const float* b_tg = (const float*)d_in[7];
    const float* w_rep= (const float*)d_in[8];  const float* b_rep= (const float*)d_in[9];
    const float* pos_local  = (const float*)d_in[10];
    const float* pos_global = (const float*)d_in[11];
    const float* WI[4] = {(const float*)d_in[12], (const float*)d_in[16], (const float*)d_in[20], (const float*)d_in[24]};
    const float* BI[4] = {(const float*)d_in[13], (const float*)d_in[17], (const float*)d_in[21], (const float*)d_in[25]};
    const float* WO[4] = {(const float*)d_in[14], (const float*)d_in[18], (const float*)d_in[22], (const float*)d_in[26]};
    const float* BO[4] = {(const float*)d_in[15], (const float*)d_in[19], (const float*)d_in[23], (const float*)d_in[27]};

    char* cur = (char*)d_ws;
    auto alloc = [&](size_t bytes) -> void* {
        void* r = cur; cur += (bytes + 255) & ~(size_t)255; return r;
    };

    bf16* w_tl_b  = (bf16*)alloc(E2 * 2);
    bf16* w_tg_b  = (bf16*)alloc(E2 * 2);
    bf16* w_rep_b = (bf16*)alloc(E2 * 2);
    bf16* wi_b[4]; bf16* wo_b[4];
    for (int i = 0; i < 4; i++) { wi_b[i] = (bf16*)alloc(3 * E2 * 2); wo_b[i] = (bf16*)alloc(E2 * 2); }
    bf16* textn_b   = (bf16*)alloc(E2 * 2);
    bf16* localn_b  = (bf16*)alloc(E2 * 2);
    bf16* klocal_b  = (bf16*)alloc(E2 * 2);
    bf16* kglobal_b = (bf16*)alloc(E2 * 2);
    float* tl_f = (float*)alloc(E2 * 4); bf16* tl_b = (bf16*)alloc(E2 * 2);
    float* tg_f = (float*)alloc(E2 * 4); bf16* tg_b = (bf16*)alloc(E2 * 2);
    float* tr_f = (float*)alloc(E2 * 4); bf16* tr_b = (bf16*)alloc(E2 * 2);
    float* lt_f = (float*)alloc(E2 * 4); bf16* lt_b = (bf16*)alloc(E2 * 2);
    float* gt_f = (float*)alloc(E2 * 4); bf16* gt_b = (bf16*)alloc(E2 * 2);
    float* ff_f = (float*)alloc(E2 * 4); bf16* ff_b = (bf16*)alloc(E2 * 2);
    bf16* rt_b  = (bf16*)alloc(E2 * 2);
    bf16*  qp_b    = (bf16*)alloc(E2 * 2);
    bf16*  kp_b    = (bf16*)alloc(E2 * 2);
    bf16*  vp_b    = (bf16*)alloc(E2 * 2);
    float* scores  = (float*)alloc(NHEAD * E2 * 4);
    bf16*  attn_b  = (bf16*)alloc(NHEAD * E2 * 2);
    bf16*  attno_b = (bf16*)alloc(E2 * 2);
    float* full_f   = (float*)alloc(E2 * 4);
    bf16*  fulln_b  = (bf16*)alloc(E2 * 2);

    auto conv = [&](const float* s, bf16* dst, long n) {
        f2b_k<<<dim3((unsigned)((n + 255) / 256)), dim3(256), 0, stream>>>(s, dst, (int)n);
    };
    auto gemmNT = [&](const bf16* A, long sA, const bf16* B, long sB,
                      float* Cf, bf16* Cb, long sC, int ldc,
                      const float* bias, const float* resid,
                      int M, int N, int K, int batch, float alpha) {
        gemm_nt<<<dim3(M / 64, N / 256, batch), dim3(256), 0, stream>>>(
            A, sA, EDIM, B, sB, EDIM, Cf, Cb, sC, ldc, bias, resid, K, alpha);
    };

    // weight conversion
    conv(w_tl, w_tl_b, E2); conv(w_tg, w_tg_b, E2); conv(w_rep, w_rep_b, E2);
    for (int i = 0; i < 4; i++) { conv(WI[i], wi_b[i], 3 * E2); conv(WO[i], wo_b[i], E2); }

    // l2norm + positional embeds
    prep<<<dim3(EDIM), dim3(256), 0, stream>>>(local_feat, global_feat, text_feat,
                                               pos_local, pos_global,
                                               localn_b, klocal_b, kglobal_b, textn_b);

    auto runMHA = [&](const bf16* q_b, const bf16* kv_b, const bf16* wib, const float* bip,
                      const bf16* wob, const float* bop, const float* resid,
                      float* out_f, bf16* out_b) {
        gemmNT(q_b, 0,  wib,            0, nullptr, qp_b, 0, EDIM, bip,            nullptr, EDIM, EDIM, EDIM, 1, 1.0f);
        gemmNT(kv_b, 0, wib + E2,       0, nullptr, kp_b, 0, EDIM, bip + EDIM,     nullptr, EDIM, EDIM, EDIM, 1, 1.0f);
        gemmNT(kv_b, 0, wib + 2 * E2,   0, nullptr, vp_b, 0, EDIM, bip + 2 * EDIM, nullptr, EDIM, EDIM, EDIM, 1, 1.0f);
        // per-head scores: q_h @ k_h^T * d^-0.5 (d=256 -> 1/16)
        gemmNT(qp_b, DHEAD, kp_b, DHEAD, scores, nullptr, E2, EDIM, nullptr, nullptr, EDIM, EDIM, DHEAD, NHEAD, 0.0625f);
        softmax_rows<<<dim3(NHEAD * EDIM), dim3(256), 0, stream>>>(scores, attn_b, EDIM);
        // attn @ V (heads write disjoint 256-col strips of attno)
        gemm_nn<<<dim3(EDIM / 64, DHEAD / 256, NHEAD), dim3(256), 0, stream>>>(
            attn_b, E2, EDIM, vp_b, DHEAD, EDIM, nullptr, attno_b, DHEAD, EDIM, EDIM, 1.0f);
        // out-proj + bias + residual
        gemmNT(attno_b, 0, wob, 0, out_f, out_b, 0, EDIM, bop, resid, EDIM, EDIM, EDIM, 1, 1.0f);
    };

    // textual_local
    gemmNT(textn_b, 0, w_tl_b, 0, tl_f, tl_b, 0, EDIM, b_tl, nullptr, EDIM, EDIM, EDIM, 1, 1.0f);
    runMHA(tl_b, klocal_b, wi_b[0], BI[0], wo_b[0], BO[0], tl_f, lt_f, lt_b);
    // textual_global
    gemmNT(textn_b, 0, w_tg_b, 0, tg_f, tg_b, 0, EDIM, b_tg, nullptr, EDIM, EDIM, EDIM, 1, 1.0f);
    runMHA(tg_b, kglobal_b, wi_b[1], BI[1], wo_b[1], BO[1], tg_f, gt_f, gt_b);
    // full_fusion (q = lt, kv = gt)
    runMHA(lt_b, gt_b, wi_b[2], BI[2], wo_b[2], BO[2], lt_f, ff_f, ff_b);
    // repeat_text
    gemmNT(textn_b, 0, w_rep_b, 0, tr_f, tr_b, 0, EDIM, b_rep, nullptr, EDIM, EDIM, EDIM, 1, 1.0f);
    runMHA(tr_b, ff_b, wi_b[3], BI[3], wo_b[3], BO[3], nullptr, nullptr, rt_b);

    // full = rt @ t_r^T
    gemmNT(rt_b, 0, tr_b, 0, full_f, nullptr, 0, EDIM, nullptr, nullptr, EDIM, EDIM, EDIM, 1, 1.0f);
    rownorm<<<dim3(EDIM), dim3(256), 0, stream>>>(full_f, fulln_b);

    // logits[g] = fulln[g-rows] @ localn[g-rows]^T  (G=4 of 256x256x1024)
    gemmNT(fulln_b, (long)DHEAD * EDIM, localn_b, (long)DHEAD * EDIM,
           (float*)d_out, nullptr, (long)DHEAD * DHEAD, DHEAD,
           nullptr, nullptr, DHEAD, DHEAD, EDIM, 4, 1.0f);
}
